// Multimodal_29918742184220
// MI455X (gfx1250) — compile-verified, matrix operand order checked by reading, not run on hardware
//
#include <hip/hip_runtime.h>
#include <math.h>

typedef __attribute__((ext_vector_type(16))) _Float16 v16h;
typedef __attribute__((ext_vector_type(8)))  _Float16 v8h;
typedef __attribute__((ext_vector_type(8)))  float    v8f;

#define DEVFN static __device__ __forceinline__

// D = A(16x32 f16) * B(32x16 f16) + C(16x16 f32)
#define WMMA_F16(A, B, C) \
  __builtin_amdgcn_wmma_f32_16x16x32_f16(false, (A), false, (B), (short)0, (C), false, false)

DEVFN v16h cat8(v8h lo, v8h hi) {
  return __builtin_shufflevector(lo, hi, 0,1,2,3,4,5,6,7,8,9,10,11,12,13,14,15);
}
// Per-lane A fragment when the lane's 32 K-values are CONTIGUOUS at rowk[0..31]:
// lane layout (§7.12.2): halfs 0..7 = k hs..hs+7, halfs 8..15 = k 16+hs..23+hs, hs = (lane>>4)*8
template <typename PT>
DEVFN v16h load_row16(const PT* rowk) {
  const int hs = ((threadIdx.x & 31) >> 4) << 3;
  v8h lo = *(const v8h*)(rowk + hs);
  v8h hi = *(const v8h*)(rowk + 16 + hs);
  return cat8(lo, hi);
}
// B fragment from a pre-swizzled panel: panel[lane][16 halfs]
DEVFN v16h load_sw16(const _Float16* panel) {
  const _Float16* p = panel + ((threadIdx.x & 31) << 4);
  v8h lo = *(const v8h*)p;
  v8h hi = *(const v8h*)(p + 8);
  return cat8(lo, hi);
}

DEVFN float sigmoidf(float x) { return 1.0f / (1.0f + expf(-x)); }

// ---------------- problem constants ----------------
#define BB   16
#define SS   60
#define NN   50
#define HID  128
#define NIMG 800
#define FCIN 15360

// ---------------- workspace byte offsets (16B aligned) ----------------
#define WIH_OFF   0u          // swizzled [24][2][32][16] f16  = 49152 B
#define WHH_OFF   49152u      // swizzled [24][4][32][16] f16  = 98304 B
#define C2B_OFF   147456u     // swizzled [4][9][32][16]  f16  = 36864 B
#define FC1B_OFF  184320u     // swizzled [4][480][32][16] f16 = 1966080 B
#define P2_OFF    2150400u    // pooled2 [800][15360] f16      = 24576000 B
#define FEATS_OFF 26726400u   // [800][64] f32
#define CNNF_OFF  26931200u   // [16][128] f32
#define NUMF_OFF  26939392u   // [16][50] f32

// ---------------- LDS layouts ----------------
#define IMG_OFF_L   0u        // padded img f32 [66][62] = 16368 -> 16384
#define C1W_OFF_L   16384u    // 288 f32
#define C1B_OFF_L   17536u    // 32 f32
#define C2BI_OFF_L  17664u    // 64 f32
#define P1_OFF_L    17920u    // padded pooled1 f16 [34][32][32] = 69632
#define C2BSW_OFF_L 87552u    // swizzled conv2 B f16 = 36864
#define C2_OFF_L    124416u   // conv2 relu out f16 [64][960] = 122880
#define CNN_LDS     247296u
#define GRU_LDS     63488u

// decompose swizzled register slot (lane, j) -> matrix k
DEVFN int sw_k(int lane, int j) {
  int hs = ((lane >> 4) << 3);
  return (j < 8) ? (hs + j) : (16 + hs + (j - 8));
}

// ============================================================
// 0) Build f16 WMMA-swizzled B panels in workspace
//    layout: [tile][kstep][lane][16 halfs]
// ============================================================
__global__ void prep_kernel(const float* __restrict__ W_ih, const float* __restrict__ W_hh,
                            const float* __restrict__ conv2_w, const float* __restrict__ fc1_w,
                            char* __restrict__ ws) {
  _Float16* wih = (_Float16*)(ws + WIH_OFF);
  _Float16* whh = (_Float16*)(ws + WHH_OFF);
  _Float16* c2B = (_Float16*)(ws + C2B_OFF);
  _Float16* f1B = (_Float16*)(ws + FC1B_OFF);
  const int gid = blockIdx.x * blockDim.x + threadIdx.x;
  const int nt  = gridDim.x * blockDim.x;
  // W_ih: 24 tiles x 2 ksteps (K padded 50->64)
  for (int i = gid; i < 24*2*32*16; i += nt) {
    int j = i & 15, lane = (i >> 4) & 31, ks = (i >> 9) & 1, ntl = i >> 10;
    int n = lane & 15, kf = ks*32 + sw_k(lane, j), g = ntl*16 + n;
    wih[i] = (kf < NN) ? (_Float16)W_ih[g*NN + kf] : (_Float16)0.f;
  }
  // W_hh: 24 tiles x 4 ksteps (K=128)
  for (int i = gid; i < 24*4*32*16; i += nt) {
    int j = i & 15, lane = (i >> 4) & 31, ks = (i >> 9) & 3, ntl = i >> 11;
    int n = lane & 15, kf = ks*32 + sw_k(lane, j), g = ntl*16 + n;
    whh[i] = (_Float16)W_hh[g*HID + kf];
  }
  // conv2: 4 tiles x 9 ksteps; K reordered: kstep = tap r, klocal = ic
  for (int i = gid; i < 4*9*32*16; i += nt) {
    int j = i & 15, lane = (i >> 4) & 31, t = i >> 9;
    int ks = t % 9, ntl = t / 9;
    int n = lane & 15, ic = sw_k(lane, j), oc = ntl*16 + n;
    c2B[i] = (_Float16)conv2_w[oc*288 + ic*9 + ks];
  }
  // fc1: 4 tiles x 480 ksteps (K=15360, natural order)
  for (int i = gid; i < 4*480*32*16; i += nt) {
    int j = i & 15, lane = (i >> 4) & 31, t = i >> 9;
    int ks = t % 480, ntl = t / 480;
    int n = lane & 15, kf = ks*32 + sw_k(lane, j), oc = ntl*16 + n;
    f1B[i] = (_Float16)fc1_w[(size_t)oc*FCIN + kf];
  }
}

// ============================================================
// 1) Fused conv1+pool -> conv2(WMMA)+pool, one block per image
// ============================================================
__global__ void cnn_kernel(const float* __restrict__ x_img,
                           const float* __restrict__ c1w_g, const float* __restrict__ c1b_g,
                           const float* __restrict__ c2b_g,
                           const _Float16* __restrict__ c2B_g,
                           _Float16* __restrict__ p2_g) {
  extern __shared__ char smem[];
  float*    imgp  = (float*)(smem + IMG_OFF_L);     // [66][62] zero halo
  float*    c1w   = (float*)(smem + C1W_OFF_L);
  float*    c1b   = (float*)(smem + C1B_OFF_L);
  float*    c2b   = (float*)(smem + C2BI_OFF_L);
  _Float16* p1    = (_Float16*)(smem + P1_OFF_L);   // [34][32][32] ch-last, zero halo
  _Float16* c2Bsw = (_Float16*)(smem + C2BSW_OFF_L);
  _Float16* c2    = (_Float16*)(smem + C2_OFF_L);   // [64][960]

  const int tid = threadIdx.x;
  const float* src = x_img + (size_t)blockIdx.x * 3840;
  for (int i = tid; i < 66*62; i += 256) {          // padded image load, no later branches
    int y = i / 62 - 1, x = i % 62 - 1;
    imgp[i] = ((unsigned)y < 64u && (unsigned)x < 60u) ? src[y*60 + x] : 0.f;
  }
  for (int i = tid; i < 288; i += 256) c1w[i] = c1w_g[i];
  if (tid < 32) c1b[tid] = c1b_g[tid];
  if (tid < 64) c2b[tid] = c2b_g[tid];
  for (int i = tid; i < 9216; i += 256)             // conv2 B panels (u32 copy)
    ((unsigned int*)c2Bsw)[i] = ((const unsigned int*)c2B_g)[i];
  for (int i = tid; i < 34*32*32/2; i += 256)       // zero p1 (halo stays 0)
    ((unsigned int*)p1)[i] = 0u;
  __syncthreads();

  // conv1(1->32) + relu + maxpool2 -> p1[(py+1)][(px+1)][c], branch-free
  for (int o = tid; o < 32*32*30; o += 256) {
    int c = o / 960, rem = o - c*960;
    int py = rem / 30, px = rem - py*30;
    const float* wr = c1w + c*9;
    float bias = c1b[c];
    float mx = -1e30f;
#pragma unroll
    for (int dy = 0; dy < 2; ++dy)
#pragma unroll
      for (int dx = 0; dx < 2; ++dx) {
        int y = 2*py + dy, x = 2*px + dx;
        float s = bias;
#pragma unroll
        for (int ky = 0; ky < 3; ++ky)
#pragma unroll
          for (int kx = 0; kx < 3; ++kx)
            s += imgp[(y+ky)*62 + (x+kx)] * wr[ky*3 + kx];
        mx = fmaxf(mx, s);
      }
    p1[((py+1)*32 + (px+1))*32 + c] = (_Float16)fmaxf(mx, 0.f);
  }
  __syncthreads();

  // conv2(32->64) implicit GEMM: M=960 px, K=9 taps x 32 ic, N=64
  const int wave = tid >> 5, lane = tid & 31, pl = lane & 15;
  for (int job = wave; job < 240; job += 8) {       // 60 M-tiles x 4 N-tiles
    int mt = job >> 2, ntl = job & 3;
    int pbase = mt * 16;
    int p = pbase + pl;
    int y = p / 30, x = p - y*30;                   // per-lane, once per job
    v8f acc = {};
#pragma unroll
    for (int ks = 0; ks < 9; ++ks) {                // tap fixed per kstep
      const int ky = ks / 3, kx = ks - ky*3;
      v16h A  = load_row16(p1 + ((y + ky)*32 + (x + kx))*32);  // 2x ds_load_b128
      v16h Bm = load_sw16(c2Bsw + ((ntl*9 + ks) << 9));        // 2x ds_load_b128
      acc = WMMA_F16(A, Bm, acc);
    }
    int oc = ntl*16 + pl;
    float bias = c2b[oc];
    v8h dv;
#pragma unroll
    for (int r = 0; r < 8; ++r) dv[r] = (_Float16)fmaxf(acc[r] + bias, 0.f);
    *(v8h*)(c2 + oc*960 + pbase + ((lane >> 4) << 3)) = dv;    // 1x ds_store_b128
  }
  __syncthreads();

  // maxpool2 -> pooled2 flat [64][16][15] f16 to workspace
  _Float16* dst = p2_g + (size_t)blockIdx.x * FCIN;
  for (int o = tid; o < FCIN; o += 256) {
    int c = o / 240, rem = o - c*240;
    int py = rem / 15, px = rem - py*15;
    float mx = -1e30f;
#pragma unroll
    for (int dy = 0; dy < 2; ++dy)
#pragma unroll
      for (int dx = 0; dx < 2; ++dx)
        mx = fmaxf(mx, (float)c2[c*960 + (2*py+dy)*30 + (2*px+dx)]);
    dst[o] = (_Float16)mx;
  }
}

// ============================================================
// 2) fc1 GEMM: [800 x 15360] x [15360 x 64], WMMA, relu
// ============================================================
__global__ void fc1_kernel(const _Float16* __restrict__ p2,
                           const _Float16* __restrict__ fc1Bsw,
                           const float* __restrict__ fc1_b,
                           float* __restrict__ feats) {
  const int wave = threadIdx.x >> 5, lane = threadIdx.x & 31;
  const int gw = blockIdx.x * 8 + wave;
  if (gw >= 200) return;                           // 50 M-tiles x 4 N-tiles
  const int mt = gw >> 2, ntl = gw & 3;
  const _Float16* arow = p2 + (size_t)(mt*16 + (lane & 15)) * FCIN;
  const _Float16* bpan = fc1Bsw + (size_t)ntl * 480 * 512;
  v8f acc = {};
  for (int ks = 0; ks < 480; ++ks) {
    v16h A  = load_row16(arow + ks*32);            // 2x global b128
    v16h Bm = load_sw16(bpan + ks*512);            // 2x global b128 (L2-resident)
    acc = WMMA_F16(A, Bm, acc);
  }
  int oc = ntl*16 + (lane & 15);
  float bia = fc1_b[oc];
#pragma unroll
  for (int r = 0; r < 8; ++r) {
    int m = r + ((lane >> 4) << 3);
    feats[(size_t)(mt*16 + m)*64 + oc] = fmaxf(acc[r] + bia, 0.f);
  }
}

// ============================================================
// 3) fc_binary + sigmoid, fc2 + mean over stocks (one block/b)
// ============================================================
__global__ void head_kernel(const float* __restrict__ feats,
                            const float* __restrict__ fc2_w, const float* __restrict__ fc2_b,
                            const float* __restrict__ fbw, const float* __restrict__ fbb,
                            float* __restrict__ cnnf, float* __restrict__ out) {
  __shared__ float sf[64];
  const int b = blockIdx.x, t = threadIdx.x;
  if (t < 64) {
    float s = 0.f;
    for (int n = 0; n < NN; ++n) s += feats[(size_t)(b*NN + n)*64 + t];
    sf[t] = s;
  }
  __syncthreads();
  if (t < 128) {
    float s = 0.f;
    for (int k = 0; k < 64; ++k) s += sf[k] * fc2_w[t*64 + k];
    cnnf[b*128 + t] = s * (1.f/50.f) + fc2_b[t];
  }
  if (t < NN) {
    int i = b*NN + t;
    float s = fbb[0];
    for (int k = 0; k < 64; ++k) s += feats[(size_t)i*64 + k] * fbw[k];
    out[BB*NN + i] = sigmoidf(s);
  }
}

// ============================================================
// 4) GRU, 60 sequential steps; gates via WMMA (M=16,N=384)
// ============================================================
__global__ void gru_kernel(const float* __restrict__ x_num,
                           const _Float16* __restrict__ Wih, const _Float16* __restrict__ Whh,
                           const float* __restrict__ b_ih, const float* __restrict__ b_hh,
                           const float* __restrict__ gfw, const float* __restrict__ gfb,
                           float* __restrict__ numf) {
  extern __shared__ char smem[];
  _Float16* xt = (_Float16*)smem;              // [16][64]
  _Float16* hh = (_Float16*)(smem + 2048);     // [16][128]
  float*    hf = (float*)(smem + 6144);        // [16][128]
  float*    gi = (float*)(smem + 14336);       // [16][384]
  float*    gh = (float*)(smem + 38912);       // [16][384]
  const int tid = threadIdx.x, wave = tid >> 5, lane = tid & 31;
  for (int i = tid; i < BB*HID; i += 256) { hf[i] = 0.f; hh[i] = (_Float16)0.f; }
  __syncthreads();
  for (int t = 0; t < SS; ++t) {
    for (int i = tid; i < BB*64; i += 256) {
      int b = i >> 6, k = i & 63;
      xt[i] = (k < NN) ? (_Float16)x_num[b*SS*NN + t*NN + k] : (_Float16)0.f;
    }
    __syncthreads();
    const _Float16* xrow = xt + (lane & 15)*64;
    const _Float16* hrow = hh + (lane & 15)*HID;
    for (int ntl = wave; ntl < 24; ntl += 8) {
      v8f ai = {}, ah = {};
#pragma unroll
      for (int ks = 0; ks < 2; ++ks)
        ai = WMMA_F16(load_row16(xrow + ks*32), load_sw16(Wih + (ntl*2 + ks)*512), ai);
#pragma unroll
      for (int ks = 0; ks < 4; ++ks)
        ah = WMMA_F16(load_row16(hrow + ks*32), load_sw16(Whh + (ntl*4 + ks)*512), ah);
      int g = ntl*16 + (lane & 15);
      float bi = b_ih[g], bh2 = b_hh[g];
#pragma unroll
      for (int r = 0; r < 8; ++r) {
        int m = r + ((lane >> 4) << 3);
        gi[m*384 + g] = ai[r] + bi;
        gh[m*384 + g] = ah[r] + bh2;
      }
    }
    __syncthreads();
    for (int i = tid; i < BB*HID; i += 256) {
      int b = i >> 7, j = i & 127, base = b*384;
      float r = sigmoidf(gi[base + j]       + gh[base + j]);
      float z = sigmoidf(gi[base + 128 + j] + gh[base + 128 + j]);
      float n = tanhf(gi[base + 256 + j] + r * gh[base + 256 + j]);
      float hn = (1.f - z) * n + z * hf[i];
      hf[i] = hn;
      hh[i] = (_Float16)hn;
    }
    __syncthreads();
  }
  for (int i = tid; i < BB*NN; i += 256) {
    int b = i / NN, o = i - b*NN;
    float s = gfb[o];
    for (int k = 0; k < HID; ++k) s += hf[b*HID + k] * gfw[o*HID + k];
    numf[i] = s;
  }
}

// ============================================================
// 5) fusion -> softmax -> 32-iter water-filling (one block/b)
// ============================================================
__global__ void fuse_kernel(const float* __restrict__ numf, const float* __restrict__ cnnf,
                            const float* __restrict__ fusion_w, const float* __restrict__ fusion_b,
                            const float* __restrict__ final_w, const float* __restrict__ final_b,
                            float* __restrict__ out) {
  __shared__ float iv[NN + 128];
  __shared__ float fu[64];
  __shared__ float lg[NN];
  __shared__ float oldw[NN], wcw[NN];
  const int b = blockIdx.x, t = threadIdx.x;       // 64 threads
  if (t < NN) iv[t] = numf[b*NN + t];
  for (int j = t; j < 128; j += 64) iv[NN + j] = cnnf[b*128 + j];
  __syncthreads();
  {
    float s = fusion_b[t];
    for (int k = 0; k < NN + 128; ++k) s += iv[k] * fusion_w[t*(NN+128) + k];
    fu[t] = fmaxf(s, 0.f);
  }
  __syncthreads();
  if (t < NN) {
    float s = final_b[t];
    for (int k = 0; k < 64; ++k) s += fu[k] * final_w[t*64 + k];
    lg[t] = s;
  }
  __syncthreads();
  if (t < NN) {
    float m = -1e30f;
    for (int k = 0; k < NN; ++k) m = fmaxf(m, lg[k]);
    float ssum = 0.f;
    for (int k = 0; k < NN; ++k) ssum += expf(lg[k] - m);
    float w = expf(lg[t] - m) / ssum;
    oldw[t] = w;
    wcw[t]  = fminf(fmaxf(w, 0.0f), 0.1f);
  }
  __syncthreads();
  for (int it = 0; it < 32; ++it) {
    float leftover = 0.f, s = 0.f;
    for (int k = 0; k < NN; ++k) {
      leftover += oldw[k] - wcw[k];
      if (wcw[k] != 0.1f) s += wcw[k];
    }
    __syncthreads();
    if (t < NN) {
      float wck = wcw[t];
      float gift = ((wck != 0.1f) && (s > 0.f)) ? leftover * wck / s : 0.f;
      float no = wck + gift;
      oldw[t] = no;
      wcw[t]  = fminf(fmaxf(no, 0.0f), 0.1f);
    }
    __syncthreads();
  }
  if (t < NN) out[b*NN + t] = wcw[t];
}

// ============================================================
extern "C" void kernel_launch(void* const* d_in, const int* in_sizes, int n_in,
                              void* d_out, int out_size, void* d_ws, size_t ws_size,
                              hipStream_t stream) {
  (void)in_sizes; (void)n_in; (void)out_size; (void)ws_size;
  const float* x_num    = (const float*)d_in[0];
  const float* x_img    = (const float*)d_in[1];
  const float* conv1_w  = (const float*)d_in[2];
  const float* conv1_b  = (const float*)d_in[3];
  const float* conv2_w  = (const float*)d_in[4];
  const float* conv2_b  = (const float*)d_in[5];
  const float* fc1_w    = (const float*)d_in[6];
  const float* fc1_b    = (const float*)d_in[7];
  const float* fbw      = (const float*)d_in[8];
  const float* fbb      = (const float*)d_in[9];
  const float* fc2_w    = (const float*)d_in[10];
  const float* fc2_b    = (const float*)d_in[11];
  const float* W_ih     = (const float*)d_in[12];
  const float* W_hh     = (const float*)d_in[13];
  const float* b_ih     = (const float*)d_in[14];
  const float* b_hh     = (const float*)d_in[15];
  const float* gfw      = (const float*)d_in[16];
  const float* gfb      = (const float*)d_in[17];
  const float* fusion_w = (const float*)d_in[18];
  const float* fusion_b = (const float*)d_in[19];
  const float* final_w  = (const float*)d_in[20];
  const float* final_b  = (const float*)d_in[21];
  char*  ws  = (char*)d_ws;
  float* out = (float*)d_out;

  (void)hipFuncSetAttribute(reinterpret_cast<const void*>(&cnn_kernel),
                            hipFuncAttributeMaxDynamicSharedMemorySize, CNN_LDS);
  (void)hipFuncSetAttribute(reinterpret_cast<const void*>(&gru_kernel),
                            hipFuncAttributeMaxDynamicSharedMemorySize, GRU_LDS);

  prep_kernel<<<512, 256, 0, stream>>>(W_ih, W_hh, conv2_w, fc1_w, ws);
  cnn_kernel<<<NIMG, 256, CNN_LDS, stream>>>(x_img, conv1_w, conv1_b, conv2_b,
      (const _Float16*)(ws + C2B_OFF), (_Float16*)(ws + P2_OFF));
  fc1_kernel<<<25, 256, 0, stream>>>((const _Float16*)(ws + P2_OFF),
      (const _Float16*)(ws + FC1B_OFF), fc1_b, (float*)(ws + FEATS_OFF));
  head_kernel<<<BB, 256, 0, stream>>>((const float*)(ws + FEATS_OFF), fc2_w, fc2_b,
      fbw, fbb, (float*)(ws + CNNF_OFF), out);
  gru_kernel<<<1, 256, GRU_LDS, stream>>>(x_num,
      (const _Float16*)(ws + WIH_OFF), (const _Float16*)(ws + WHH_OFF),
      b_ih, b_hh, gfw, gfb, (float*)(ws + NUMF_OFF));
  fuse_kernel<<<BB, 64, 0, stream>>>((const float*)(ws + NUMF_OFF),
      (const float*)(ws + CNNF_OFF), fusion_w, fusion_b, final_w, final_b, out);
}
